// StepRNN_21457656611064
// MI455X (gfx1250) — matrix operand order repaired and checked by reading, not compiled, not run on hardware
//
#include <hip/hip_runtime.h>
#include <hip/hip_bf16.h>

typedef __attribute__((ext_vector_type(16))) __bf16       v16bf;
typedef __attribute__((ext_vector_type(8)))  float        v8f;
typedef __attribute__((ext_vector_type(4)))  unsigned int v4u;

#define B_   128
#define S_   512
#define I_   256
#define H_   1024
#define NREC 64   // persistent workgroups for the recurrent scan (one per 16-col slice)

// ---------------------------------------------------------------- utilities

__global__ void init_barrier_kernel(unsigned* bar) {
    if (threadIdx.x < 2) bar[threadIdx.x] = 0u;
}

__global__ void cvt_f32_bf16_kernel(const float* __restrict__ src,
                                    __bf16* __restrict__ dst, int n) {
    int i      = blockIdx.x * blockDim.x + threadIdx.x;
    int stride = gridDim.x * blockDim.x;
    for (; i < n; i += stride) dst[i] = (__bf16)src[i];
}

// Device-wide split barrier: release-fence my stores, arrive, spin on generation,
// then ALL threads acquire-fence (L0 invalidate) before reading peer data.
__device__ __forceinline__ void grid_barrier(unsigned* cnt, unsigned* gen, unsigned nb) {
    __syncthreads();
    if (threadIdx.x == 0) {
        __builtin_amdgcn_fence(__ATOMIC_RELEASE, "agent");       // global_wb of my h stores
        unsigned g    = __hip_atomic_load(gen, __ATOMIC_RELAXED, __HIP_MEMORY_SCOPE_AGENT);
        unsigned prev = __hip_atomic_fetch_add(cnt, 1u, __ATOMIC_ACQ_REL, __HIP_MEMORY_SCOPE_AGENT);
        if (prev == nb - 1u) {
            __hip_atomic_store(cnt, 0u, __ATOMIC_RELAXED, __HIP_MEMORY_SCOPE_AGENT);
            __hip_atomic_store(gen, g + 1u, __ATOMIC_RELEASE, __HIP_MEMORY_SCOPE_AGENT);
        } else {
            while (__hip_atomic_load(gen, __ATOMIC_ACQUIRE, __HIP_MEMORY_SCOPE_AGENT) == g) {
                __builtin_amdgcn_s_sleep(1);
            }
        }
    }
    __syncthreads();
    __builtin_amdgcn_fence(__ATOMIC_ACQUIRE, "agent");           // every wave invalidates
}

// ---------------------------------------------------------------- x_proj GEMM
// XP[s][b][h] = sum_i X[b][s][i] * W_ih[h][i] + b_ih[h] + b_hh[h]
// M = S*B = 65536 (m = s*128 + b), N = H = 1024, K = I = 256.
// Each wave computes one M-tile x TWO adjacent N-tiles (A-fragment reuse x2).
// 8 waves/block -> block covers 128 rows x 32 cols. grid = 512 * 32 = 16384 blocks.

__global__ __launch_bounds__(256) void xproj_gemm_kernel(
    const __bf16* __restrict__ Xbf,    // [B][S][I]
    const __bf16* __restrict__ Wih,    // [H][I]
    const float*  __restrict__ b_ih,
    const float*  __restrict__ b_hh,
    float*        __restrict__ XP)     // [S][B][H]
{
    const int wave = threadIdx.x >> 5;
    const int lane = threadIdx.x & 31;
    const int lo   = lane & 15;
    const int hi   = lane >> 4;

    const int np = blockIdx.x & 31;                 // 32 N-tile pairs
    const int tm = ((blockIdx.x >> 5) << 3) + wave; // 4096 M-tiles
    const int m0 = tm << 4;
    const int n0 = np << 5;                         // two N-tiles: n0, n0+16
    const int s  = m0 >> 7;                         // 16 | 128, so one s per tile
    const int b0 = m0 & 127;

    const __bf16* arow  = Xbf + ((size_t)(b0 + lo) * S_ + s) * I_;   // A row for this lane
    const __bf16* brow0 = Wih + (size_t)(n0 + lo) * I_;              // B^T rows for this lane
    const __bf16* brow1 = Wih + (size_t)(n0 + 16 + lo) * I_;

    v8f acc0 = {}, acc1 = {};
    #pragma unroll
    for (int k0 = 0; k0 < I_; k0 += 32) {
        union { v16bf v; v4u u[2]; } a, b0f, b1f;
        // A 16x32 bf16: lanes 0-15 K=0..15 ; lanes 16-31 K=16..31
        a.u[0]   = *(const v4u*)(arow + k0 + hi * 16);
        a.u[1]   = *(const v4u*)(arow + k0 + hi * 16 + 8);
        // B 32x16 bf16 (from B^T row-major): lanes 0-15 K=0..15 ; lanes 16-31 K=16..31
        b0f.u[0] = *(const v4u*)(brow0 + k0 + hi * 16);
        b0f.u[1] = *(const v4u*)(brow0 + k0 + hi * 16 + 8);
        b1f.u[0] = *(const v4u*)(brow1 + k0 + hi * 16);
        b1f.u[1] = *(const v4u*)(brow1 + k0 + hi * 16 + 8);
        acc0 = __builtin_amdgcn_wmma_f32_16x16x32_bf16(false, a.v, false, b0f.v,
                                                       (short)0, acc0, false, false);
        acc1 = __builtin_amdgcn_wmma_f32_16x16x32_bf16(false, a.v, false, b1f.v,
                                                       (short)0, acc1, false, false);
    }

    const float bias0 = b_ih[n0 + lo]      + b_hh[n0 + lo];
    const float bias1 = b_ih[n0 + 16 + lo] + b_hh[n0 + 16 + lo];
    float* out = XP + ((size_t)s * B_ + b0) * H_ + n0;
    #pragma unroll
    for (int i = 0; i < 8; ++i) {
        const int m = i + hi * 8;               // row within tile (C layout)
        out[(size_t)m * H_ + lo]      = acc0[i] + bias0;
        out[(size_t)m * H_ + 16 + lo] = acc1[i] + bias1;
    }
}

// ---------------------------------------------------------------- persistent recurrent scan
// 64 workgroups x 256 threads. Workgroup j owns columns [16j,16j+16) and keeps that
// 16x1024 slice of W_hh in LDS for all 512 steps. Wave w owns M-tile w (rows 16w..16w+15).
// Per step: acc = h_t . W_hh^T (slice), h_{t+1} = tanh(acc + x_proj[t]); fp32 -> HS[t],
// bf16 -> ping-pong buffer for the next step's A operand. Device barrier between steps.

__global__ __launch_bounds__(256) void rnn_scan_kernel(
    const __bf16* __restrict__ Whh,   // [H][H] bf16
    __bf16*       __restrict__ hbuf,  // [2][B][H] bf16 ping-pong (buf0 = h0)
    float*        __restrict__ HS,    // [S][B][H]: holds x_proj, overwritten with h_t
    unsigned*     __restrict__ bar)
{
    __shared__ __attribute__((aligned(16))) __bf16 Wsl[16 * H_];   // 32 KB

    const int tid  = threadIdx.x;
    const int wave = tid >> 5;
    const int lane = tid & 31;
    const int lo   = lane & 15;
    const int hi   = lane >> 4;
    const int n0   = blockIdx.x << 4;   // this WG's column slice
    const int m0   = wave << 4;         // this wave's row tile
    const unsigned nb = gridDim.x;

    // Stage W_hh rows [n0, n0+16) into LDS (contiguous 32 KB in row-major W_hh).
    {
        const v4u* src = (const v4u*)(Whh + (size_t)n0 * H_);
        v4u*       dst = (v4u*)Wsl;
        #pragma unroll
        for (int i = 0; i < (16 * H_ / 8) / 256; ++i)
            dst[tid + i * 256] = src[tid + i * 256];
    }
    __syncthreads();

    const __bf16* brow = Wsl + (size_t)lo * H_;   // B^T row (local) for this lane

    for (int t = 0; t < S_; ++t) {
        const __bf16* cur = hbuf + (size_t)(t & 1) * (B_ * H_);
        __bf16*       nxt = hbuf + (size_t)((t + 1) & 1) * (B_ * H_);
        float*        xp  = HS + (size_t)t * (B_ * H_);

        // Prefetch next step's x_proj tile while we do this step's math.
        if (t + 1 < S_) {
            const float* pf = HS + (size_t)(t + 1) * (B_ * H_)
                                 + (size_t)(m0 + hi * 8) * H_ + n0 + lo;
            __builtin_prefetch(pf, 0, 1);
        }

        const __bf16* arow = cur + (size_t)(m0 + lo) * H_;
        v8f acc = {};
        #pragma unroll 4
        for (int k0 = 0; k0 < H_; k0 += 32) {
            union { v16bf v; v4u u[2]; } a, bb;
            a.u[0]  = *(const v4u*)(arow + k0 + hi * 16);
            a.u[1]  = *(const v4u*)(arow + k0 + hi * 16 + 8);
            bb.u[0] = *(const v4u*)(brow + k0 + hi * 16);
            bb.u[1] = *(const v4u*)(brow + k0 + hi * 16 + 8);
            acc = __builtin_amdgcn_wmma_f32_16x16x32_bf16(false, a.v, false, bb.v,
                                                          (short)0, acc, false, false);
        }

        #pragma unroll
        for (int i = 0; i < 8; ++i) {
            const int    m   = m0 + i + hi * 8;
            const size_t idx = (size_t)m * H_ + n0 + lo;
            const float  hv  = tanhf(acc[i] + xp[idx]);
            xp[idx]  = hv;            // HS[t] := h_{t+1} (fp32 output)
            nxt[idx] = (__bf16)hv;    // operand for next step
        }

        grid_barrier(bar, bar + 1, nb);
    }
}

// ---------------------------------------------------------------- epilogue kernels

__global__ __launch_bounds__(256) void fc_out_kernel(
    const float* __restrict__ hfin,   // HS[S-1] : [B][H]
    const float* __restrict__ Wfc,    // [H]
    const float* __restrict__ bfc,    // [1]
    float*       __restrict__ out)    // [B]
{
    __shared__ float red[256];
    const int b = blockIdx.x;
    float s = 0.f;
    for (int i = threadIdx.x; i < H_; i += 256)
        s += hfin[(size_t)b * H_ + i] * Wfc[i];
    red[threadIdx.x] = s;
    __syncthreads();
    for (int off = 128; off > 0; off >>= 1) {
        if (threadIdx.x < off) red[threadIdx.x] += red[threadIdx.x + off];
        __syncthreads();
    }
    if (threadIdx.x == 0) out[b] = red[0] + bfc[0];
}

__global__ void copy_states_kernel(const float* __restrict__ HS,
                                   float* __restrict__ ss, float* __restrict__ sf) {
    const int n      = B_ * H_;
    int       i      = blockIdx.x * blockDim.x + threadIdx.x;
    const int stride = gridDim.x * blockDim.x;
    for (; i < n; i += stride) {
        ss[i] = HS[i];                            // hs[0]
        sf[i] = HS[(size_t)(S_ - 1) * n + i];     // hs[S-1] = h_final
    }
}

// ---------------------------------------------------------------- launch

extern "C" void kernel_launch(void* const* d_in, const int* in_sizes, int n_in,
                              void* d_out, int out_size, void* d_ws, size_t ws_size,
                              hipStream_t stream) {
    const float* X      = (const float*)d_in[0];  // [B][S][I]
    const float* hidden = (const float*)d_in[1];  // [B][1][H]
    const float* W_ih   = (const float*)d_in[2];  // [H][I]
    const float* W_hh   = (const float*)d_in[3];  // [H][H]
    const float* b_ih   = (const float*)d_in[4];  // [H]
    const float* b_hh   = (const float*)d_in[5];  // [H]
    const float* W_fc   = (const float*)d_in[6];  // [1][H]
    const float* b_fc   = (const float*)d_in[7];  // [1]

    // d_out layout: out(128) | hs(S*B*H) | state_start(B*H) | state_final(B*H)
    float* out = (float*)d_out;
    float* HS  = out + 128;
    float* ss  = HS + (size_t)S_ * B_ * H_;
    float* sf  = ss + (size_t)B_ * H_;

    // workspace layout (bytes)
    unsigned char* ws   = (unsigned char*)d_ws;
    unsigned*      bar  = (unsigned*)ws;                          // 256 B
    __bf16*        Xbf  = (__bf16*)(ws + 256);                    // 33,554,432 B
    __bf16*        Wih  = (__bf16*)(ws + 256 + 33554432ull);      //    524,288 B
    __bf16*        Whh  = (__bf16*)(ws + 256 + 33554432ull + 524288ull);              // 2,097,152 B
    __bf16*        hbuf = (__bf16*)(ws + 256 + 33554432ull + 524288ull + 2097152ull); //   524,288 B

    init_barrier_kernel<<<1, 32, 0, stream>>>(bar);
    cvt_f32_bf16_kernel<<<4096, 256, 0, stream>>>(X,      Xbf,  B_ * S_ * I_);
    cvt_f32_bf16_kernel<<<256,  256, 0, stream>>>(W_ih,   Wih,  H_ * I_);
    cvt_f32_bf16_kernel<<<512,  256, 0, stream>>>(W_hh,   Whh,  H_ * H_);
    cvt_f32_bf16_kernel<<<64,   256, 0, stream>>>(hidden, hbuf, B_ * H_);   // h0 -> ping buf 0

    // x_proj -> HS region of d_out (each wave: 1 M-tile x 2 N-tiles)
    xproj_gemm_kernel<<<(S_ * B_ / 16 / 8) * (H_ / 32), 256, 0, stream>>>(
        Xbf, Wih, b_ih, b_hh, HS);

    // persistent sequential scan (overwrites HS[t] with h_{t+1})
    rnn_scan_kernel<<<NREC, 256, 0, stream>>>(Whh, hbuf, HS, bar);

    copy_states_kernel<<<256, 256, 0, stream>>>(HS, ss, sf);
    fc_out_kernel<<<B_, 256, 0, stream>>>(HS + (size_t)(S_ - 1) * B_ * H_, W_fc, b_fc, out);
}
